// SparseAutoencoder_31396210933964
// MI455X (gfx1250) — compile-verified, matrix operand order checked by reading, not run on hardware
//
#include <hip/hip_runtime.h>
#include <hip/hip_bf16.h>

// ---------------------------------------------------------------------------
// TopK Sparse Autoencoder forward for MI455X (gfx1250, wave32, WMMA).
//
// Inputs (fp32): x[16,512,768], w_enc[768,24576], w_dec[24576,768],
//                b_enc[24576], b_pre[768]
// Output (fp32): recons[16,512,768]
//
// Encoder GEMM (309 GFLOP) dominates -> v_wmma_f32_16x16x32_bf16 with fp32
// accumulate. w_enc converted to bf16 [H][D] (37.7 MB, L2-resident: 192 MB).
// pre_acts (805 MB) round-trips HBM (~70us @ 23.3 TB/s) to keep top-k exact.
// A tiles staged to LDS via GLOBAL_LOAD_ASYNC_TO_LDS_B128 (ASYNCcnt-tracked
// LDS-DMA); falls back to classic load+ds_store if the builtin is absent.
// ---------------------------------------------------------------------------

#define TOK_TOTAL 8192
#define D_DIM     768
#define H_DIM     24576
#define K_SEL     64
#define BM        64      // tokens per workgroup in GEMM
#define BN        128     // h-columns per workgroup (8 waves * 16)
#define LDA       776     // A row stride in LDS (bf16 elems), +16B pad: bank-conflict-free

typedef __attribute__((ext_vector_type(16))) __bf16 v16bf;
typedef __attribute__((ext_vector_type(8)))  float  v8f;
typedef __attribute__((ext_vector_type(4)))  int    v4i;

typedef __attribute__((address_space(1))) v4i* gptr_v4i;   // global int4*
typedef __attribute__((address_space(3))) v4i* lptr_v4i;   // LDS int4*

union FragBF {           // 16 bf16 = 8 VGPRs, loaded as two 16-byte chunks
  v16bf v;
  uint4 q[2];
};

#if defined(__HIP_DEVICE_COMPILE__) && defined(__gfx1250__) && \
    __has_builtin(__builtin_amdgcn_global_load_async_to_lds_b128)
#define USE_ASYNC_LDS 1
#else
#define USE_ASYNC_LDS 0
#endif

// ------------------------------- Kernel 1: LayerNorm -> bf16 ----------------
__global__ __launch_bounds__(256) void k_layernorm(
    const float* __restrict__ x, const float* __restrict__ b_pre,
    __hip_bfloat16* __restrict__ xnbf, float* __restrict__ mu_out,
    float* __restrict__ sd_out)
{
  __shared__ float scratch[9];
  const int tid  = threadIdx.x;
  const int lane = tid & 31;
  const int tok  = blockIdx.x;
  const float* xr = x + (size_t)tok * D_DIM;

  float v0 = xr[tid], v1 = xr[tid + 256], v2 = xr[tid + 512];

  // block sum -> mean
  float s = v0 + v1 + v2;
  for (int off = 16; off > 0; off >>= 1) s += __shfl_down(s, off);
  if (lane == 0) scratch[tid >> 5] = s;
  __syncthreads();
  if (tid == 0) { float t = 0.f; for (int i = 0; i < 8; ++i) t += scratch[i]; scratch[8] = t; }
  __syncthreads();
  const float mu = scratch[8] * (1.0f / (float)D_DIM);

  const float c0 = v0 - mu, c1 = v1 - mu, c2 = v2 - mu;
  float ss = c0 * c0 + c1 * c1 + c2 * c2;
  for (int off = 16; off > 0; off >>= 1) ss += __shfl_down(ss, off);
  if (lane == 0) scratch[tid >> 5] = ss;
  __syncthreads();
  if (tid == 0) { float t = 0.f; for (int i = 0; i < 8; ++i) t += scratch[i]; scratch[8] = t; }
  __syncthreads();
  const float sdv = sqrtf(scratch[8] * (1.0f / (float)(D_DIM - 1)));  // ddof=1
  const float inv = 1.0f / (sdv + 1e-5f);

  __hip_bfloat16* o = xnbf + (size_t)tok * D_DIM;
  o[tid]       = __float2bfloat16(c0 * inv - b_pre[tid]);
  o[tid + 256] = __float2bfloat16(c1 * inv - b_pre[tid + 256]);
  o[tid + 512] = __float2bfloat16(c2 * inv - b_pre[tid + 512]);
  if (tid == 0) { mu_out[tok] = mu; sd_out[tok] = sdv; }
}

// ---------------- Kernel 2: w_enc [D][H] fp32 -> bf16 transposed [H][D] -----
__global__ __launch_bounds__(256) void k_wenc_cvt(
    const float* __restrict__ w_enc, __hip_bfloat16* __restrict__ wtb)
{
  __shared__ float tile[32][33];
  const int hb = blockIdx.x * 32;   // along H
  const int db = blockIdx.y * 32;   // along D
  const int c  = threadIdx.x & 31;
  const int r  = threadIdx.x >> 5;  // 8 rows per pass
#pragma unroll
  for (int rr = 0; rr < 32; rr += 8)
    tile[r + rr][c] = w_enc[(size_t)(db + r + rr) * H_DIM + hb + c];
  __syncthreads();
#pragma unroll
  for (int rr = 0; rr < 32; rr += 8)
    wtb[(size_t)(hb + r + rr) * D_DIM + db + c] = __float2bfloat16(tile[c][r + rr]);
}

// ---------------- Kernel 3: encoder GEMM (bf16 WMMA) + b_enc ----------------
// Grid: (TOK_TOTAL/BM, H_DIM/BN), 256 threads (8 waves).
// Wave w owns h-stripe [hBase + w*16, +16) and all 4 token-subtiles of 16.
__global__ __launch_bounds__(256) void k_encoder_gemm(
    const __hip_bfloat16* __restrict__ xnbf, const __hip_bfloat16* __restrict__ wtb,
    const float* __restrict__ b_enc, float* __restrict__ preacts)
{
  extern __shared__ __align__(16) __hip_bfloat16 Asm[];  // [BM][LDA] bf16
  const int tid  = threadIdx.x;
  const int lane = tid & 31;
  const int wave = tid >> 5;
  const int tokBase = blockIdx.x * BM;
  const int hBase   = blockIdx.y * BN + wave * 16;

  // Stage the 64x768 bf16 A tile into LDS (rows padded to LDA).
  const __hip_bfloat16* Ag = xnbf + (size_t)tokBase * D_DIM;
#if USE_ASYNC_LDS
  for (int i = tid; i < BM * (D_DIM / 8); i += 256) {
    const int row = i / (D_DIM / 8);
    const int c8  = i % (D_DIM / 8);
    __builtin_amdgcn_global_load_async_to_lds_b128(
        (gptr_v4i)(Ag + (size_t)row * D_DIM + c8 * 8),
        (lptr_v4i)(&Asm[row * LDA + c8 * 8]),
        0, 0);
  }
#if __has_builtin(__builtin_amdgcn_s_wait_asynccnt)
  __builtin_amdgcn_s_wait_asynccnt(0);
#else
  asm volatile("s_wait_asynccnt 0x0" ::: "memory");
#endif
#else
  for (int i = tid; i < BM * (D_DIM / 8); i += 256) {
    const int row = i / (D_DIM / 8);
    const int c8  = i % (D_DIM / 8);
    *(uint4*)(&Asm[row * LDA + c8 * 8]) = *(const uint4*)(Ag + (size_t)row * D_DIM + c8 * 8);
  }
#endif
  __syncthreads();

  // ISA 16-bit fragment layout: lane<16 -> K{0..7,16..23}, lane>=16 -> K{8..15,24..31}
  const int half = lane >> 4;
  const int r    = lane & 15;
  const __hip_bfloat16* Bg = wtb + (size_t)(hBase + r) * D_DIM;  // B column r == row of wtb

  v8f acc[4] = {};
  for (int k0 = 0; k0 < D_DIM; k0 += 32) {
    FragBF b;
    b.q[0] = *(const uint4*)(Bg + k0 + half * 8);
    b.q[1] = *(const uint4*)(Bg + k0 + 16 + half * 8);
    if (k0 + 32 < D_DIM)
      __builtin_prefetch(Bg + k0 + 32, 0, 3);   // global_prefetch, near-cache locality
#pragma unroll
    for (int t = 0; t < 4; ++t) {
      FragBF a;
      const __hip_bfloat16* Ar = &Asm[(t * 16 + r) * LDA + k0];
      a.q[0] = *(const uint4*)(Ar + half * 8);
      a.q[1] = *(const uint4*)(Ar + 16 + half * 8);
      acc[t] = __builtin_amdgcn_wmma_f32_16x16x32_bf16(
          /*neg_a=*/false, a.v, /*neg_b=*/false, b.v,
          /*c_mod=*/(short)0, acc[t], /*reuse_a=*/false, /*reuse_b=*/false);
    }
  }

  // C/D layout: VGPR j -> M=j (lanes 0-15) / M=j+8 (lanes 16-31), N=lane%16.
  const int h = hBase + r;
  const float be = b_enc[h];
#pragma unroll
  for (int t = 0; t < 4; ++t) {
#pragma unroll
    for (int j = 0; j < 8; ++j) {
      const int tokLocal = t * 16 + j + half * 8;
      preacts[(size_t)(tokBase + tokLocal) * H_DIM + h] = acc[t][j] + be;
    }
  }
}

// ---------------- Kernel 4: exact per-token top-64 (+ReLU) ------------------
// One workgroup per token; row lives in LDS (96 KB of the 320 KB WGP pool).
__global__ __launch_bounds__(256) void k_topk(
    const float* __restrict__ preacts, float* __restrict__ tkv, int* __restrict__ tki)
{
  extern __shared__ __align__(16) float rowbuf[];  // H_DIM floats
  __shared__ float redv[8];
  __shared__ int   redi[8];
  const int tid  = threadIdx.x;
  const int lane = tid & 31;
  const int wave = tid >> 5;
  const int tok  = blockIdx.x;
  const float* src = preacts + (size_t)tok * H_DIM;

  for (int i = tid; i < H_DIM / 4; i += 256)
    ((float4*)rowbuf)[i] = ((const float4*)src)[i];
  __syncthreads();

  for (int sel = 0; sel < K_SEL; ++sel) {
    float bv = -__builtin_inff();
    int   bi = 0x7fffffff;
    for (int i = tid; i < H_DIM; i += 256) {
      const float v = rowbuf[i];
      if (v > bv || (v == bv && i < bi)) { bv = v; bi = i; }
    }
    for (int off = 16; off > 0; off >>= 1) {      // wave32 argmax reduce
      const float ov = __shfl_down(bv, off);
      const int   oi = __shfl_down(bi, off);
      if (ov > bv || (ov == bv && oi < bi)) { bv = ov; bi = oi; }
    }
    if (lane == 0) { redv[wave] = bv; redi[wave] = bi; }
    __syncthreads();
    if (tid == 0) {
      for (int w = 1; w < 8; ++w) {
        const float ov = redv[w]; const int oi = redi[w];
        if (ov > bv || (ov == bv && oi < bi)) { bv = ov; bi = oi; }
      }
      tkv[(size_t)tok * K_SEL + sel] = bv > 0.f ? bv : 0.f;   // relu(topk)
      tki[(size_t)tok * K_SEL + sel] = bi;
      rowbuf[bi] = -__builtin_inff();                          // remove winner
    }
    __syncthreads();
  }
}

// ---------------- Kernel 5: sparse decoder + de-normalization ---------------
__global__ __launch_bounds__(256) void k_decode(
    const float* __restrict__ tkv, const int* __restrict__ tki,
    const float* __restrict__ w_dec, const float* __restrict__ b_pre,
    const float* __restrict__ mu, const float* __restrict__ sd,
    float* __restrict__ out)
{
  __shared__ float sval[K_SEL];
  __shared__ int   sidx[K_SEL];
  const int tid = threadIdx.x;
  const int tok = blockIdx.x;
  if (tid < K_SEL) {
    sval[tid] = tkv[(size_t)tok * K_SEL + tid];
    sidx[tid] = tki[(size_t)tok * K_SEL + tid];
  }
  __syncthreads();
  const float m = mu[tok], s = sd[tok];
  for (int d = tid; d < D_DIM; d += 256) {
    float acc = b_pre[d];
#pragma unroll 4
    for (int j = 0; j < K_SEL; ++j)
      acc = fmaf(sval[j], w_dec[(size_t)sidx[j] * D_DIM + d], acc);
    out[(size_t)tok * D_DIM + d] = acc * s + m;   // recons * std + mu
  }
}

// ------------------------------- launch -------------------------------------
extern "C" void kernel_launch(void* const* d_in, const int* in_sizes, int n_in,
                              void* d_out, int out_size, void* d_ws, size_t ws_size,
                              hipStream_t stream)
{
  const float* x     = (const float*)d_in[0];
  const float* w_enc = (const float*)d_in[1];
  const float* w_dec = (const float*)d_in[2];
  const float* b_enc = (const float*)d_in[3];
  const float* b_pre = (const float*)d_in[4];
  float* out = (float*)d_out;

  char* ws = (char*)d_ws;
  size_t off = 0;
  auto take = [&](size_t bytes) -> void* {
    void* p = ws + off;
    off += (bytes + 255) & ~(size_t)255;
    return p;
  };
  __hip_bfloat16* xnbf = (__hip_bfloat16*)take((size_t)TOK_TOTAL * D_DIM * 2);
  __hip_bfloat16* wtb  = (__hip_bfloat16*)take((size_t)H_DIM * D_DIM * 2);
  float* mu   = (float*)take((size_t)TOK_TOTAL * 4);
  float* sd   = (float*)take((size_t)TOK_TOTAL * 4);
  float* tkv  = (float*)take((size_t)TOK_TOTAL * K_SEL * 4);
  int*   tki  = (int*)  take((size_t)TOK_TOTAL * K_SEL * 4);
  float* preacts = (float*)take((size_t)TOK_TOTAL * H_DIM * 4);  // 805 MB

  k_layernorm<<<TOK_TOTAL, 256, 0, stream>>>(x, b_pre, xnbf, mu, sd);
  k_wenc_cvt<<<dim3(H_DIM / 32, D_DIM / 32), 256, 0, stream>>>(w_enc, wtb);
  k_encoder_gemm<<<dim3(TOK_TOTAL / BM, H_DIM / BN), 256,
                   (size_t)BM * LDA * 2, stream>>>(xnbf, wtb, b_enc, preacts);
  k_topk<<<TOK_TOTAL, 256, (size_t)H_DIM * 4, stream>>>(preacts, tkv, tki);
  k_decode<<<TOK_TOTAL, 256, 0, stream>>>(tkv, tki, w_dec, b_pre, mu, sd, out);
}
